// GAE_25993142075429
// MI455X (gfx1250) — matrix operand order repaired and checked by reading, not compiled, not run on hardware
//
#include <hip/hip_runtime.h>

// GCN autoencoder forward for MI455X (gfx1250, wave32).
// GEMMs via v_wmma_f32_16x16x32_f16; edge scatter via L2-resident f32 atomics.

typedef __attribute__((ext_vector_type(16))) _Float16 v16h;
typedef __attribute__((ext_vector_type(8)))  float    v8f;

__global__ __launch_bounds__(256) void k_fill(float* p, float v, int n) {
  int i = blockIdx.x * 256 + threadIdx.x;
  if (i < n) p[i] = v;
}

__global__ __launch_bounds__(256) void k_count_deg(const long long* __restrict__ tgt,
                                                   float* deg, int E) {
  int i = blockIdx.x * 256 + threadIdx.x;
  if (i < E) atomicAdd(&deg[(int)tgt[i]], 1.0f);
}

__global__ __launch_bounds__(256) void k_dinv(const float* __restrict__ deg,
                                              float* __restrict__ dinv, int n) {
  int i = blockIdx.x * 256 + threadIdx.x;
  if (i < n) dinv[i] = rsqrtf(deg[i]);  // deg >= 1 (self-loop), no zero guard needed
}

// Pack W[K x NCOLS] (row-major f32) into per-lane WMMA B-fragments (f16).
// Fragment (kb, nt), lane L: element e holds W[kb*32 + 8*(L/16) + e + (e>=8 ? 8 : 0)][nt*16 + L%16].
__global__ __launch_bounds__(256) void k_pack(const float* __restrict__ W,
                                              _Float16* __restrict__ out,
                                              int KB, int NT, int NCOLS) {
  int total = KB * NT * 32;
  for (int idx = blockIdx.x * blockDim.x + threadIdx.x; idx < total;
       idx += gridDim.x * blockDim.x) {
    int lane = idx & 31;
    int t    = idx >> 5;
    int nt   = t % NT;
    int kb   = t / NT;
    int ncol = (lane & 15) + nt * 16;
    int g    = lane >> 4;
#pragma unroll
    for (int e = 0; e < 16; ++e) {
      int k = kb * 32 + 8 * g + e + (e >= 8 ? 8 : 0);
      out[(size_t)idx * 16 + e] = (_Float16)W[(size_t)k * NCOLS + ncol];
    }
  }
}

// Y[nrows x NCOLS] = A[nrows x KDIM] * W (packed fragments), f16 WMMA, f32 accum.
// One wave per 16-row strip; NT accumulator tiles; K unrolled in steps of 32.
template <int KDIM, int NCOLS>
__global__ __launch_bounds__(256) void k_gemm_wmma(const float* __restrict__ A,
                                                   const _Float16* __restrict__ PW,
                                                   float* __restrict__ Y, int nrows) {
  constexpr int KB = KDIM / 32;
  constexpr int NT = NCOLS / 16;
  const int lane    = threadIdx.x & 31;
  const int strip   = blockIdx.x * 8 + (threadIdx.x >> 5);   // wave-uniform
  const int nStrips = (nrows + 15) >> 4;
  if (strip >= nStrips) return;                              // uniform branch: EXEC stays all-1s
  const int row0 = strip << 4;
  const int m = lane & 15, g = lane >> 4;
  int rA = row0 + m;
  if (rA > nrows - 1) rA = nrows - 1;                        // clamp loads; stores predicated
  const float* arow = A + (size_t)rA * KDIM + 8 * g;

  v8f zero = {};
  v8f acc[NT];
#pragma unroll
  for (int t = 0; t < NT; ++t) acc[t] = zero;

#pragma unroll
  for (int kb = 0; kb < KB; ++kb) {
    const float* xp = arow + kb * 32;
    v16h a;  // A layout: e<8 -> K=8g+e ; e>=8 -> K=16+8g+(e-8)
#pragma unroll
    for (int j = 0; j < 8; ++j) {
      a[j]     = (_Float16)xp[j];
      a[8 + j] = (_Float16)xp[16 + j];
    }
#pragma unroll
    for (int t = 0; t < NT; ++t) {
      v16h b = *((const v16h*)PW + (kb * NT + t) * 32 + lane);  // 32B vector load
      acc[t] = __builtin_amdgcn_wmma_f32_16x16x32_f16(
          false, a, false, b, (short)0, acc[t], false, false);
    }
  }
  // C/D layout: VGPR r -> row m = r + 8g, col = lane%16
#pragma unroll
  for (int t = 0; t < NT; ++t) {
#pragma unroll
    for (int r = 0; r < 8; ++r) {
      int row = row0 + r + 8 * g;
      if (row < nrows) Y[(size_t)row * NCOLS + t * 16 + m] = acc[t][r];
    }
  }
}

// out[idx] = v[idx] * dinv[i]^2   (self-loop message; also initializes the accumulator)
__global__ __launch_bounds__(256) void k_self(const float* __restrict__ v,
                                              const float* __restrict__ dinv,
                                              float* __restrict__ out,
                                              int total, int shift) {
  int idx = blockIdx.x * 256 + threadIdx.x;
  if (idx >= total) return;
  int i = idx >> shift;
  float di = dinv[i];
  out[idx] = v[idx] * di * di;
}

// AGG[tgt] += H[src] * dinv[src]*dinv[tgt];  one thread per (edge, 4-dim chunk)
template <int LOGQ>  // DIM = 4 << LOGQ
__global__ __launch_bounds__(256) void k_edge_scatter(const long long* __restrict__ src,
                                                      const long long* __restrict__ tgt,
                                                      const float* __restrict__ dinv,
                                                      const float* __restrict__ H,
                                                      float* __restrict__ AGG, int E) {
  constexpr int DIM = 4 << LOGQ;
  int idx = blockIdx.x * 256 + threadIdx.x;
  int e = idx >> LOGQ;
  if (e >= E) return;
  int q = idx & ((1 << LOGQ) - 1);
  int s = (int)src[e], t = (int)tgt[e];
  float w = dinv[s] * dinv[t];
  const float4 v = *reinterpret_cast<const float4*>(H + (size_t)s * DIM + q * 4);
  float* dst = AGG + (size_t)t * DIM + q * 4;
  atomicAdd(dst + 0, v.x * w);
  atomicAdd(dst + 1, v.y * w);
  atomicAdd(dst + 2, v.z * w);
  atomicAdd(dst + 3, v.w * w);
}

__global__ __launch_bounds__(256) void k_bias_act(float* __restrict__ buf,
                                                  const float* __restrict__ bias,
                                                  int total, int mask, int relu) {
  int idx = blockIdx.x * 256 + threadIdx.x;
  if (idx >= total) return;
  float v = buf[idx] + bias[idx & mask];
  buf[idx] = relu ? fmaxf(v, 0.0f) : v;
}

// recon[e] = dot(z[src], z[tgt]) over 32 dims; one wave per edge, coalesced rows.
__global__ __launch_bounds__(256) void k_decode(const float* __restrict__ z,
                                                const long long* __restrict__ src,
                                                const long long* __restrict__ tgt,
                                                float* __restrict__ recon, int E) {
  int lane = threadIdx.x & 31;
  int e = blockIdx.x * 8 + (threadIdx.x >> 5);
  if (e >= E) return;
  int s = (int)src[e], t = (int)tgt[e];
  float p = z[(size_t)s * 32 + lane] * z[(size_t)t * 32 + lane];
#pragma unroll
  for (int off = 16; off; off >>= 1) p += __shfl_xor(p, off, 32);
  if (lane == 0) recon[e] = p;
}

extern "C" void kernel_launch(void* const* d_in, const int* in_sizes, int n_in,
                              void* d_out, int out_size, void* d_ws, size_t ws_size,
                              hipStream_t stream) {
  const float* x  = (const float*)d_in[0];
  const float* W1 = (const float*)d_in[1];
  const float* b1 = (const float*)d_in[2];
  const float* W2 = (const float*)d_in[3];
  const float* b2 = (const float*)d_in[4];
  const long long* ei = (const long long*)d_in[5];

  const int n = in_sizes[0] / 128;  // nodes
  const int E = in_sizes[5] / 2;    // edges
  const long long* src = ei;
  const long long* tgt = ei + E;

  // workspace carve-out (256B aligned)
  size_t off = 0;
  auto alloc = [&](size_t bytes) -> void* {
    void* p = (char*)d_ws + off;
    off += (bytes + 255) & ~(size_t)255;
    return p;
  };
  float*     deg  = (float*)alloc(sizeof(float) * n);
  float*     dinv = (float*)alloc(sizeof(float) * n);
  float*     HX   = (float*)alloc(sizeof(float) * (size_t)n * 64);  // X@W1, later reused as Z1
  float*     AGG  = (float*)alloc(sizeof(float) * (size_t)n * 64);  // agg1 -> H (in place)
  _Float16*  PW1  = (_Float16*)alloc(sizeof(_Float16) * 4 * 4 * 32 * 16);
  _Float16*  PW2  = (_Float16*)alloc(sizeof(_Float16) * 2 * 2 * 32 * 16);
  float*     Z1   = HX;  // n*32, HX dead by then

  float* zout  = (float*)d_out;             // [n,32]
  float* recon = zout + (size_t)n * 32;     // [E]

  auto cdiv = [](int a, int b) { return (a + b - 1) / b; };

  // degrees + symmetric norm
  k_fill<<<cdiv(n, 256), 256, 0, stream>>>(deg, 1.0f, n);
  k_count_deg<<<cdiv(E, 256), 256, 0, stream>>>(tgt, deg, E);
  k_dinv<<<cdiv(n, 256), 256, 0, stream>>>(deg, dinv, n);

  // pack weights into WMMA B-fragment layout (f16)
  k_pack<<<2, 256, 0, stream>>>(W1, PW1, 4, 4, 64);
  k_pack<<<1, 256, 0, stream>>>(W2, PW2, 2, 2, 32);

  // layer 1: HX = X@W1 ; agg = D^-1/2 A_hat D^-1/2 HX ; H = relu(agg + b1)
  const int strips = cdiv(n, 16);
  k_gemm_wmma<128, 64><<<cdiv(strips, 8), 256, 0, stream>>>(x, PW1, HX, n);
  k_self<<<cdiv(n * 64, 256), 256, 0, stream>>>(HX, dinv, AGG, n * 64, 6);
  k_edge_scatter<4><<<cdiv(E * 16, 256), 256, 0, stream>>>(src, tgt, dinv, HX, AGG, E);
  k_bias_act<<<cdiv(n * 64, 256), 256, 0, stream>>>(AGG, b1, n * 64, 63, 1);

  // layer 2: Z1 = H@W2 ; z = D^-1/2 A_hat D^-1/2 Z1 + b2  (accumulated in d_out)
  k_gemm_wmma<64, 32><<<cdiv(strips, 8), 256, 0, stream>>>(AGG, PW2, Z1, n);
  k_self<<<cdiv(n * 32, 256), 256, 0, stream>>>(Z1, dinv, zout, n * 32, 5);
  k_edge_scatter<3><<<cdiv(E * 8, 256), 256, 0, stream>>>(src, tgt, dinv, Z1, zout, E);
  k_bias_act<<<cdiv(n * 32, 256), 256, 0, stream>>>(zout, b2, n * 32, 31, 0);

  // decode: per-edge dot product of embeddings
  k_decode<<<cdiv(E, 8), 256, 0, stream>>>(zout, src, tgt, recon, E);
}
